// Loss_Labels_44487271252810
// MI455X (gfx1250) — compile-verified, matrix operand order checked by reading, not run on hardware
//
#include <hip/hip_runtime.h>

// CDNA5 wave32 WMMA types
typedef __attribute__((ext_vector_type(16))) _Float16 v16h;
typedef __attribute__((ext_vector_type(8)))  float    v8f;

// ---------------------------------------------------------------------------
// Kernel 1: zero the workspace accumulator (d_ws is NOT re-poisoned between
// timed replays, so we must clear it every call).
// ---------------------------------------------------------------------------
__global__ void zero_ws_kernel(float* __restrict__ ws) {
    if (blockIdx.x == 0 && threadIdx.x == 0) ws[0] = 0.0f;
}

// ---------------------------------------------------------------------------
// Kernel 2: streaming compare + WMMA in-wave reduction.
//
// Hot loop is branchless: the host passes the number of FULL chunks
// (16 elements per lane per chunk), so there are no per-lane bounds checks
// and EXEC is all-ones at every WMMA by construction. Each chunk:
//   - 8 coalesced global_load_b128 (4 from each array)
//   - pack (ant > syn) as f16 {1,0} into the A tile
//   - one v_wmma_f32_16x16x32_f16 against an all-ones B folds the wave's
//     512 predicates into the f32 C accumulator tile.
// Every C column replicates its row-sum, so butterfly(sum of C regs) equals
// 16 * wave_count; the 1/16 scale is exact and commutes with the butterfly.
// Remainder elements (none for B = 8M) go through a scalar tail loop.
// All addends are exact dyadic rationals in f32 -> the per-wave float
// atomics are order-independent and deterministic.
// ---------------------------------------------------------------------------
__global__ __launch_bounds__(256) void count_gt_kernel(
        const float4* __restrict__ syn4,
        const float4* __restrict__ ant4,
        float* __restrict__ ws,
        int n_chunks,     // full 16-elem-per-lane chunks
        int tail_start,   // first element of the tail (= n_chunks * 16 * T)
        int n)            // total elements
{
    const int tid = blockIdx.x * blockDim.x + threadIdx.x;
    const int T   = gridDim.x * blockDim.x;

    const _Float16 h1 = (_Float16)1.0f;
    const _Float16 h0 = (_Float16)0.0f;

    v16h ones;
#pragma unroll
    for (int i = 0; i < 16; ++i) ones[i] = h1;

    v8f c = {};  // f32 C/D accumulator tile (8 VGPRs)

    for (int chunk = 0; chunk < n_chunks; ++chunk) {
        const int base = chunk * 4 * T + tid;   // float4 index, 32-bit math
        float4 s[4], t[4];
#pragma unroll
        for (int j = 0; j < 4; ++j) {           // 8x global_load_b128, coalesced
            s[j] = syn4[base + j * T];
            t[j] = ant4[base + j * T];
        }
        v16h a;
#pragma unroll
        for (int j = 0; j < 4; ++j) {
            a[4 * j + 0] = (t[j].x > s[j].x) ? h1 : h0;
            a[4 * j + 1] = (t[j].y > s[j].y) ? h1 : h0;
            a[4 * j + 2] = (t[j].z > s[j].z) ? h1 : h0;
            a[4 * j + 3] = (t[j].w > s[j].w) ? h1 : h0;
        }
        // C += A x ones : each C element accumulates a 32-wide row-sum of A.
        c = __builtin_amdgcn_wmma_f32_16x16x32_f16(
                /*neg_a=*/false, a, /*neg_b=*/false, ones,
                /*c_mod=*/(short)0, c, /*reuse_a=*/false, /*reuse_b=*/false);
    }

    // Scalar tail (empty when 16*T divides n, as it does for B = 8M).
    float extra = 0.0f;
    {
        const float* __restrict__ sf = (const float*)syn4;
        const float* __restrict__ tf = (const float*)ant4;
        for (int i = tail_start + tid; i < n; i += T)
            extra += (tf[i] > sf[i]) ? 1.0f : 0.0f;
    }

    // Per-lane partial; 1/16 undoes the 16-column replication of the C tile
    // (exact power of two, commutes with the linear butterfly).
    float p = (c[0] + c[1] + c[2] + c[3] + c[4] + c[5] + c[6] + c[7]) * 0.0625f
            + extra;

    // wave32 butterfly -> every lane holds the wave's total count.
#pragma unroll
    for (int m = 16; m >= 1; m >>= 1) p += __shfl_xor(p, m, 32);

    if ((threadIdx.x & 31) == 0) atomicAdd(ws, p);
}

// ---------------------------------------------------------------------------
// Kernel 3: finalize  out = -count / n
// ---------------------------------------------------------------------------
__global__ void finalize_kernel(const float* __restrict__ ws,
                                float* __restrict__ out,
                                float inv_n) {
    if (blockIdx.x == 0 && threadIdx.x == 0) out[0] = -(ws[0] * inv_n);
}

// ---------------------------------------------------------------------------
extern "C" void kernel_launch(void* const* d_in, const int* in_sizes, int n_in,
                              void* d_out, int out_size, void* d_ws, size_t ws_size,
                              hipStream_t stream) {
    // in[0] = S1_out (only defines batch size -> never read: saves 32 MB of HBM)
    // in[1] = synonymy_score, in[2] = antonymy_score
    const float4* syn4 = (const float4*)d_in[1];
    const float4* ant4 = (const float4*)d_in[2];
    float* out = (float*)d_out;
    float* ws  = (float*)d_ws;
    const int n = in_sizes[1];

    // 1024 blocks x 256 threads = 8192 wave32s: enough parallelism to saturate
    // 23.3 TB/s on a 64 MB stream (~2.75 us roofline).
    const int block = 256;
    const int grid  = 1024;
    const int T     = grid * block;           // 262144 threads
    const int n_chunks   = n / (16 * T);      // full branchless chunks
    const int tail_start = n_chunks * 16 * T; // == n for B = 8M (no tail)

    zero_ws_kernel<<<1, 64, 0, stream>>>(ws);
    count_gt_kernel<<<grid, block, 0, stream>>>(syn4, ant4, ws,
                                                n_chunks, tail_start, n);
    finalize_kernel<<<1, 64, 0, stream>>>(ws, out, 1.0f / (float)n);
}
